// Attention_34265249087699
// MI455X (gfx1250) — compile-verified, matrix operand order checked by reading, not run on hardware
//
#include <hip/hip_runtime.h>

// ---------------------------------------------------------------------------
// MHA forward for MI455X (gfx1250): bf16 WMMA + async global->LDS staging.
// B=2, N=2048, C=1024, H=16, D=64.
// ---------------------------------------------------------------------------

typedef __bf16 bf16_t;
typedef __attribute__((ext_vector_type(16))) __bf16 v16bf;
typedef __attribute__((ext_vector_type(8)))  __bf16 v8bf;
typedef __attribute__((ext_vector_type(8)))  float  v8f;

#define B_  2
#define N_  2048
#define C_  1024
#define H_  16
#define D_  64
#define BN_ (B_ * N_)
#define C3_ (3 * C_)

// Build a 16x32 A (or B^T) bf16 WMMA fragment for one lane.
// Lane holds row (lane&15); half = lane>>4 selects K chunks
// [8*half, +8) and [16+8*half, +8) within the 32-wide K window.
__device__ __forceinline__ v16bf frag_from(const bf16_t* base, int half) {
  v8bf lo = *(const v8bf*)(base + 8 * half);
  v8bf hi = *(const v8bf*)(base + 16 + 8 * half);
  return __builtin_shufflevector(lo, hi, 0, 1, 2, 3, 4, 5, 6, 7,
                                 8, 9, 10, 11, 12, 13, 14, 15);
}

__device__ __forceinline__ v8f wmma_bf16(v16bf a, v16bf b, v8f c) {
  return __builtin_amdgcn_wmma_f32_16x16x32_bf16(
      false, a, false, b, (short)0, c, false, false);
}

// Async DMA one 16-byte chunk global -> LDS (CDNA5; tracked by ASYNCcnt).
__device__ __forceinline__ void async_copy_b128(const bf16_t* gptr, bf16_t* lptr) {
  unsigned lds_addr = (unsigned)(size_t)lptr;          // generic -> LDS offset
  unsigned long long gaddr = (unsigned long long)(size_t)gptr;
  asm volatile("global_load_async_to_lds_b128 %0, %1, off"
               :: "v"(lds_addr), "v"(gaddr) : "memory");
}

__device__ __forceinline__ void wait_async0() {
  asm volatile("s_wait_asynccnt 0x0" ::: "memory");
}

// ---------------------------------------------------------------------------
// Kernel 1a: elementwise f32 -> bf16
// ---------------------------------------------------------------------------
__global__ void cvt_f32_bf16(const float* __restrict__ src,
                             bf16_t* __restrict__ dst, int n) {
  int i = blockIdx.x * blockDim.x + threadIdx.x;
  if (i < n) dst[i] = (bf16_t)src[i];
}

// ---------------------------------------------------------------------------
// Kernel 1b: tiled transpose + convert: src f32 [R, Cc] -> dst bf16 [Cc, R]
// ---------------------------------------------------------------------------
__global__ __launch_bounds__(256) void transpose_cvt(
    const float* __restrict__ src, bf16_t* __restrict__ dst, int R, int Cc) {
  __shared__ float tile[32][33];
  const int bx = blockIdx.x * 32, by = blockIdx.y * 32;
  const int tx = threadIdx.x, ty = threadIdx.y;
#pragma unroll
  for (int i = 0; i < 32; i += 8)
    tile[ty + i][tx] = src[(size_t)(by + ty + i) * Cc + bx + tx];
  __syncthreads();
#pragma unroll
  for (int i = 0; i < 32; i += 8)
    dst[(size_t)(bx + ty + i) * R + by + tx] = (bf16_t)tile[tx][ty + i];
}

// ---------------------------------------------------------------------------
// Kernel 2: QKV GEMM.  xa bf16 [BN, C] (A), wT bf16 [3C, C] (B^T).
// Workgroup tile 256x64, 8 waves x (32x64).  Double-buffered async B tiles.
// Epilogue scatters into q (scaled), k, v^T bf16 buffers.
// ---------------------------------------------------------------------------
__global__ __launch_bounds__(256) void qkv_gemm(
    const bf16_t* __restrict__ xa, const bf16_t* __restrict__ wT,
    bf16_t* __restrict__ qb, bf16_t* __restrict__ kb,
    bf16_t* __restrict__ vTb) {
  __shared__ bf16_t ldsB[2][64 * 32];  // 2 x (64 cols x 32 K) = 8 KB
  const int tid = threadIdx.x;
  const int wave = tid >> 5, lane = tid & 31;
  const int half = lane >> 4, l16 = lane & 15;
  const int m0 = blockIdx.x * 256, n0 = blockIdx.y * 64;

  const int sr = tid >> 2, sc = (tid & 3) * 8;     // staging row / chunk
  const bf16_t* sgp = &wT[(size_t)(n0 + sr) * C_ + sc];

  const bf16_t* aptr0 = xa + (size_t)(m0 + wave * 32 + l16) * C_;
  const bf16_t* aptr1 = aptr0 + (size_t)16 * C_;

  v8f acc[2][4] = {};

  async_copy_b128(sgp, &ldsB[0][sr * 32 + sc]);    // prologue: stage k0 = 0
  wait_async0();
  __syncthreads();

  for (int k0 = 0; k0 < C_; k0 += 32) {
    const int buf = (k0 >> 5) & 1;
    if (k0 + 32 < C_)
      async_copy_b128(sgp + k0 + 32, &ldsB[buf ^ 1][sr * 32 + sc]);

    v16bf a0 = frag_from(aptr0 + k0, half);
    v16bf a1 = frag_from(aptr1 + k0, half);
#pragma unroll
    for (int nt = 0; nt < 4; nt++) {
      v16bf b = frag_from(&ldsB[buf][(nt * 16 + l16) * 32], half);
      acc[0][nt] = wmma_bf16(a0, b, acc[0][nt]);
      acc[1][nt] = wmma_bf16(a1, b, acc[1][nt]);
    }
    wait_async0();       // next buffer landed (and our reads of buf are done)
    __syncthreads();
  }

  // Epilogue: C tile f32 layout: row = v + 8*half, col = l16.
#pragma unroll
  for (int str = 0; str < 2; str++) {
#pragma unroll
    for (int nt = 0; nt < 4; nt++) {
      const int col = n0 + nt * 16 + l16;   // in [0, 3C)
      const int which = col / C_;           // 0=q, 1=k, 2=v
      const int cc = col % C_;
      const int h = cc / D_, d = cc % D_;
#pragma unroll
      for (int v = 0; v < 8; v++) {
        const int grow = m0 + wave * 32 + str * 16 + v + 8 * half;
        const int b = grow / N_, n = grow % N_;
        const int bh = b * H_ + h;
        const float f = acc[str][nt][v];
        if (which == 0)
          qb[((size_t)bh * N_ + n) * D_ + d] = (bf16_t)(f * 0.125f);  // D^-0.5
        else if (which == 1)
          kb[((size_t)bh * N_ + n) * D_ + d] = (bf16_t)f;
        else
          vTb[((size_t)bh * D_ + d) * N_ + n] = (bf16_t)f;  // V transposed
      }
    }
  }
}

// ---------------------------------------------------------------------------
// Kernel 3: flash attention.  Grid (B*H, N/128), 256 threads (8 waves),
// each wave owns a 16-query strip, iterates keys in blocks of 32.
// ---------------------------------------------------------------------------
__global__ __launch_bounds__(256) void attn_kernel(
    const bf16_t* __restrict__ qb, const bf16_t* __restrict__ kb,
    const bf16_t* __restrict__ vTb, bf16_t* __restrict__ attn) {
  __shared__ bf16_t ldsP[8 * 16 * 32];  // per-wave 16x32 P staging, 8 KB
  const int tid = threadIdx.x, wave = tid >> 5, lane = tid & 31;
  const int half = lane >> 4, l16 = lane & 15;
  const int bh = blockIdx.x;
  const int b = bh / H_, h = bh % H_;
  const int q0 = blockIdx.y * 128 + wave * 16;

  const bf16_t* qrow = qb + ((size_t)bh * N_ + q0 + l16) * D_;
  const v16bf qa0 = frag_from(qrow, half);       // d = 0..31
  const v16bf qa1 = frag_from(qrow + 32, half);  // d = 32..63

  v8f acc[4] = {};
  v8f mrow, lrow;
#pragma unroll
  for (int v = 0; v < 8; v++) { mrow[v] = -3.0e38f; lrow[v] = 0.0f; }

  bf16_t* pbase = &ldsP[wave * 512];
  const bf16_t* kbase = kb + (size_t)bh * N_ * D_;
  const bf16_t* vbase = vTb + (size_t)bh * D_ * N_;

  for (int j0 = 0; j0 < N_; j0 += 32) {
    // ---- S = Q * K^T for a 16x32 score tile (two 16x16 C tiles) ----
    v8f s[2];
#pragma unroll
    for (int t = 0; t < 2; t++) {
      const bf16_t* krow = kbase + (size_t)(j0 + t * 16 + l16) * D_;
      v16bf kf0 = frag_from(krow, half);
      v16bf kf1 = frag_from(krow + 32, half);
      v8f c = {};
      c = wmma_bf16(qa0, kf0, c);
      c = wmma_bf16(qa1, kf1, c);
      s[t] = c;
    }

    // ---- online softmax: row stats via shfl reductions in 16-lane halves
    v8f pm;
#pragma unroll
    for (int v = 0; v < 8; v++) pm[v] = fmaxf(s[0][v], s[1][v]);
#pragma unroll
    for (int off = 8; off >= 1; off >>= 1)
#pragma unroll
      for (int v = 0; v < 8; v++) pm[v] = fmaxf(pm[v], __shfl_xor(pm[v], off));

    v8f alpha, psum;
#pragma unroll
    for (int v = 0; v < 8; v++) {
      const float mnew = fmaxf(mrow[v], pm[v]);
      alpha[v] = __expf(mrow[v] - mnew);
      mrow[v] = mnew;
      s[0][v] = __expf(s[0][v] - mnew);
      s[1][v] = __expf(s[1][v] - mnew);
      psum[v] = s[0][v] + s[1][v];
    }
#pragma unroll
    for (int off = 8; off >= 1; off >>= 1)
#pragma unroll
      for (int v = 0; v < 8; v++) psum[v] += __shfl_xor(psum[v], off);
#pragma unroll
    for (int v = 0; v < 8; v++) lrow[v] = lrow[v] * alpha[v] + psum[v];
#pragma unroll
    for (int t2 = 0; t2 < 4; t2++)
#pragma unroll
      for (int v = 0; v < 8; v++) acc[t2][v] *= alpha[v];

    // ---- C-layout -> A-layout via per-wave LDS slab ----
#pragma unroll
    for (int t = 0; t < 2; t++)
#pragma unroll
      for (int v = 0; v < 8; v++)
        pbase[(v + 8 * half) * 32 + t * 16 + l16] = (bf16_t)s[t][v];
    asm volatile("s_wait_dscnt 0" ::: "memory");  // wave-local LDS visibility

    v16bf pf = frag_from(&pbase[l16 * 32], half);

    // ---- O += P * V  (V^T rows are d, keys contiguous) ----
#pragma unroll
    for (int t2 = 0; t2 < 4; t2++) {
      const bf16_t* vrow = vbase + (size_t)(t2 * 16 + l16) * N_ + j0;
      v16bf vf = frag_from(vrow, half);
      acc[t2] = wmma_bf16(pf, vf, acc[t2]);
    }
    asm volatile("s_wait_dscnt 0" ::: "memory");  // P loads done before reuse
  }

  // ---- normalize and store to [B, N, C] bf16 (A operand of proj GEMM) ----
#pragma unroll
  for (int t2 = 0; t2 < 4; t2++)
#pragma unroll
    for (int v = 0; v < 8; v++) {
      const int qg = q0 + v + 8 * half;
      const float o = acc[t2][v] / lrow[v];
      attn[((size_t)(b * N_ + qg)) * C_ + h * D_ + t2 * 16 + l16] = (bf16_t)o;
    }
}

// ---------------------------------------------------------------------------
// Kernel 4: output projection GEMM + bias, f32 output.
// Same 256x64 block / 32x64 wave tiling with async double-buffered B.
// ---------------------------------------------------------------------------
__global__ __launch_bounds__(256) void proj_gemm(
    const bf16_t* __restrict__ xa, const bf16_t* __restrict__ wT,
    const float* __restrict__ bias, float* __restrict__ out) {
  __shared__ bf16_t ldsB[2][64 * 32];
  const int tid = threadIdx.x;
  const int wave = tid >> 5, lane = tid & 31;
  const int half = lane >> 4, l16 = lane & 15;
  const int m0 = blockIdx.x * 256, n0 = blockIdx.y * 64;

  const int sr = tid >> 2, sc = (tid & 3) * 8;
  const bf16_t* sgp = &wT[(size_t)(n0 + sr) * C_ + sc];

  const bf16_t* aptr0 = xa + (size_t)(m0 + wave * 32 + l16) * C_;
  const bf16_t* aptr1 = aptr0 + (size_t)16 * C_;

  v8f acc[2][4] = {};

  async_copy_b128(sgp, &ldsB[0][sr * 32 + sc]);
  wait_async0();
  __syncthreads();

  for (int k0 = 0; k0 < C_; k0 += 32) {
    const int buf = (k0 >> 5) & 1;
    if (k0 + 32 < C_)
      async_copy_b128(sgp + k0 + 32, &ldsB[buf ^ 1][sr * 32 + sc]);

    v16bf a0 = frag_from(aptr0 + k0, half);
    v16bf a1 = frag_from(aptr1 + k0, half);
#pragma unroll
    for (int nt = 0; nt < 4; nt++) {
      v16bf b = frag_from(&ldsB[buf][(nt * 16 + l16) * 32], half);
      acc[0][nt] = wmma_bf16(a0, b, acc[0][nt]);
      acc[1][nt] = wmma_bf16(a1, b, acc[1][nt]);
    }
    wait_async0();
    __syncthreads();
  }

#pragma unroll
  for (int str = 0; str < 2; str++) {
#pragma unroll
    for (int nt = 0; nt < 4; nt++) {
      const int col = n0 + nt * 16 + l16;
      const float bv = bias[col];
#pragma unroll
      for (int v = 0; v < 8; v++) {
        const int grow = m0 + wave * 32 + str * 16 + v + 8 * half;
        out[(size_t)grow * C_ + col] = acc[str][nt][v] + bv;
      }
    }
  }
}

// ---------------------------------------------------------------------------
// Launch
// ---------------------------------------------------------------------------
extern "C" void kernel_launch(void* const* d_in, const int* in_sizes, int n_in,
                              void* d_out, int out_size, void* d_ws,
                              size_t ws_size, hipStream_t stream) {
  (void)in_sizes; (void)n_in; (void)out_size; (void)ws_size;
  const float* x      = (const float*)d_in[0];
  const float* w_qkv  = (const float*)d_in[1];
  const float* w_proj = (const float*)d_in[2];
  const float* b_proj = (const float*)d_in[3];
  float* out = (float*)d_out;

  char* ws = (char*)d_ws;
  bf16_t* xb     = (bf16_t*)(ws + 0);                    //  8 MB  [BN, C]
  bf16_t* wqkvT  = (bf16_t*)(ws + (size_t)8  * 1048576); //  6 MB  [3C, C]
  bf16_t* wprojT = (bf16_t*)(ws + (size_t)14 * 1048576); //  2 MB  [C, C]
  bf16_t* qb     = (bf16_t*)(ws + (size_t)16 * 1048576); //  8 MB  [B,H,N,D]
  bf16_t* kb     = (bf16_t*)(ws + (size_t)24 * 1048576); //  8 MB  [B,H,N,D]
  bf16_t* vTb    = (bf16_t*)(ws + (size_t)32 * 1048576); //  8 MB  [B,H,D,N]
  bf16_t* attnb  = (bf16_t*)(ws + (size_t)40 * 1048576); //  8 MB  [BN, C]

  cvt_f32_bf16<<<(BN_ * C_ + 255) / 256, 256, 0, stream>>>(x, xb, BN_ * C_);
  dim3 tb(32, 8);
  transpose_cvt<<<dim3(C3_ / 32, C_ / 32), tb, 0, stream>>>(w_qkv, wqkvT, C_, C3_);
  transpose_cvt<<<dim3(C_ / 32, C_ / 32), tb, 0, stream>>>(w_proj, wprojT, C_, C_);

  qkv_gemm<<<dim3(BN_ / 256, C3_ / 64), 256, 0, stream>>>(xb, wqkvT, qb, kb, vTb);
  attn_kernel<<<dim3(B_ * H_, N_ / 128), 256, 0, stream>>>(qb, kb, vTb, attnb);
  proj_gemm<<<dim3(BN_ / 256, C_ / 64), 256, 0, stream>>>(attnb, wprojT, b_proj, out);
}
